// DenseTensor_78323023610453
// MI455X (gfx1250) — matrix operand order repaired and checked by reading, not compiled, not run on hardware
//
#include <hip/hip_runtime.h>
#include <hip/hip_bf16.h>

// Problem constants (match reference): B=1024, D=512, U=512
#define BN 1024
#define DN 512
#define UN 512

typedef __attribute__((ext_vector_type(16))) __bf16 v16bf;
typedef __attribute__((ext_vector_type(8)))  float  v8f;
typedef __attribute__((ext_vector_type(4)))  unsigned int v4u;
typedef __attribute__((ext_vector_type(8)))  int v8i;
typedef __attribute__((ext_vector_type(4)))  int v4i;

// ---------------------------------------------------------------------------
// Kernel 0: convert x (f32) -> bf16 workspace (1 MB, stays L2-resident)
// ---------------------------------------------------------------------------
__global__ __launch_bounds__(256)
void xq_convert_kernel(const float* __restrict__ x, __bf16* __restrict__ xq, int n) {
    int i = blockIdx.x * 256 + threadIdx.x;
    if (i < n) xq[i] = (__bf16)x[i];
}

// ---------------------------------------------------------------------------
// TDM: issue one 2D tile load  V[d0..d0+31, u, e0..e0+15] (f32) -> LDS.
// Tile: tile_dim0 = 16 elems (x, contiguous 64B rows), tile_dim1 = 32 rows,
// row stride = U*D elements. data_size = 4 bytes (code 2).
// ---------------------------------------------------------------------------
__device__ __forceinline__ void tdm_load_vtile(const float* gsrc, unsigned lds_off) {
    unsigned long long ga = (unsigned long long)(uintptr_t)gsrc;
    v4u g0;
    g0[0] = 1u;                                       // count=1 (valid user D#)
    g0[1] = lds_off;                                  // lds_addr (bytes)
    g0[2] = (unsigned)(ga & 0xFFFFFFFFu);             // global_addr[31:0]
    g0[3] = (unsigned)((ga >> 32) & 0x01FFFFFFu)      // global_addr[56:32]
          | (2u << 30);                               // type=2 ("image")
    v8i g1;
    g1[0] = (int)(2u << 16);                          // data_size = 4B
    g1[1] = (int)(512u << 16);                        // tensor_dim0 = 512 (bits 79:48 lo)
    g1[2] = 0;                                        // tensor_dim0 hi, tensor_dim1 lo16
    g1[3] = (int)((16u << 16) | 4u);                  // tile_dim0=16 | tensor_dim1 hi (262144>>16)
    g1[4] = 32;                                       // tile_dim1 = 32, tile_dim2 = 0
    g1[5] = (int)(DN * UN);                           // tensor_dim0_stride lo32 = 262144
    g1[6] = 0;                                        // stride hi, dim1_stride lo
    g1[7] = 0;
    v4i gz4  = {0, 0, 0, 0};
    v8i gz8  = {0, 0, 0, 0, 0, 0, 0, 0};
    __builtin_amdgcn_tensor_load_to_lds(g0, g1, gz4, gz4, gz8, 0);
}

// ---------------------------------------------------------------------------
// Kernel 1: fused  out[b,u] = bias[u] + sum_e x[b,e] * M'[b,e]
//           where  M'[b,e] = W[e,u] + sum_d x[b,d] * V[d,u,e]
// Grid (U, 2): block (u, gy) owns b-rows [gy*512, gy*512+512).
// 8 waves x 64 b-rows each (4 WMMA subtiles -> 32 acc VGPRs, no spills).
// V tiles streamed by the Tensor Data Mover into triple-buffered LDS.
// ---------------------------------------------------------------------------
__global__ __launch_bounds__(256)
void bilinear_wmma_kernel(const float* __restrict__ x,
                          const float* __restrict__ W,
                          const float* __restrict__ V,
                          const float* __restrict__ bias,
                          const __bf16* __restrict__ xq,
                          float* __restrict__ out)
{
    __shared__ float vtile[3][32 * 16];   // 3 x 2KB TDM buffers
    __shared__ float quad[512];           // per-b accumulator across e-tiles

    const int u    = blockIdx.x;
    const int gy   = blockIdx.y;          // which half of B
    const int tid  = threadIdx.x;
    const int wave = tid >> 5;
    const int lane = tid & 31;
    const int col  = lane & 15;
    const int hi   = lane >> 4;
    const int brow0 = gy * 512;           // this block's first global b-row

    for (int i = tid; i < 512; i += 256) quad[i] = 0.0f;
    __syncthreads();

    const float* vblk = V + (long)u * DN;             // &V[0, u, 0]
    // Base of this wave's A panel in bf16 x
    const __bf16* arow = xq + (long)(brow0 + wave * 64 + col) * DN + hi * 16;

    for (int et = 0; et < 32; ++et) {
        const int e0 = et * 16;

        // Linear term folded into accumulator init: C[m,n] = W[e0+n, u]
        const float winit = W[(long)(e0 + col) * UN + u];
        v8f acc[4];
        #pragma unroll
        for (int t = 0; t < 4; ++t) {
            #pragma unroll
            for (int j = 0; j < 8; ++j) acc[t][j] = winit;
        }

        // Prologue: DMA tiles for dit=0,1 into buffers 0,1 (wave 0 only)
        if (wave == 0) {
            tdm_load_vtile(vblk + e0,                        (unsigned)(uintptr_t)&vtile[0][0]);
            tdm_load_vtile(vblk + (long)32 * (DN * UN) + e0, (unsigned)(uintptr_t)&vtile[1][0]);
        }

        for (int dit = 0; dit < 16; ++dit) {
            const int cur = dit % 3;
            const int d0  = dit * 32;

            // Ensure tile `cur` has landed (allow the next DMA to stay in flight)
            if (wave == 0) {
                if (dit < 15) __builtin_amdgcn_s_wait_tensorcnt(1);
                else          __builtin_amdgcn_s_wait_tensorcnt(0);
            }
            __syncthreads();

            // Kick DMA for dit+2 into the buffer last read at dit-1 (fenced above)
            if (wave == 0 && dit + 2 < 16) {
                tdm_load_vtile(vblk + (long)(d0 + 64) * (DN * UN) + e0,
                               (unsigned)(uintptr_t)&vtile[(dit + 2) % 3][0]);
            }

            // Batch-issue all A-fragment loads first (one big vmem clause),
            // so WMMAs can retire against partial load counts.
            v16bf afrag[4];
            #pragma unroll
            for (int t = 0; t < 4; ++t) {
                afrag[t] = *(const v16bf*)(arow + (long)t * (16 * DN) + d0);
            }

            // B fragment (32x16, K=d, N=e): lane owns e-column col, 16 d values d=16*hi+j.
            // LDS reads + bf16 packing overlap the global A loads above.
            v16bf bfrag;
            {
                const float* row = &vtile[cur][hi * 16 * 16 + col];
                #pragma unroll
                for (int j = 0; j < 16; ++j) bfrag[j] = (__bf16)row[j * 16];
            }

            // 4 b-subtiles reuse the B fragment
            #pragma unroll
            for (int t = 0; t < 4; ++t) {
                acc[t] = __builtin_amdgcn_wmma_f32_16x16x32_bf16(
                    false, afrag[t], false, bfrag, (short)0, acc[t], false, false);
            }
        }

        // Epilogue: quad[b] += sum_{n} M'[b, e0+n] * x[b, e0+n]
        #pragma unroll
        for (int t = 0; t < 4; ++t) {
            const int bbase = wave * 64 + t * 16 + hi * 8;
            #pragma unroll
            for (int j = 0; j < 8; ++j) {
                const int bl = bbase + j;
                float p = acc[t][j] * x[(long)(brow0 + bl) * DN + e0 + col];
                p += __shfl_xor(p, 1, 32);
                p += __shfl_xor(p, 2, 32);
                p += __shfl_xor(p, 4, 32);
                p += __shfl_xor(p, 8, 32);
                if (col == 0) quad[bl] += p;   // lanes 0 and 16 -> distinct b
            }
        }
        __syncthreads();
    }

    const float bu = bias[u];
    for (int i = tid; i < 512; i += 256) {
        out[(long)(brow0 + i) * UN + u] = quad[i] + bu;
    }
}

// ---------------------------------------------------------------------------
// Harness entry
// ---------------------------------------------------------------------------
extern "C" void kernel_launch(void* const* d_in, const int* in_sizes, int n_in,
                              void* d_out, int out_size, void* d_ws, size_t ws_size,
                              hipStream_t stream) {
    const float* x    = (const float*)d_in[0];   // (B, D)
    const float* W    = (const float*)d_in[1];   // (D, U)
    const float* V    = (const float*)d_in[2];   // (D, U, D)
    const float* bias = (const float*)d_in[3];   // (U,)
    float* out = (float*)d_out;                  // (B, U)
    __bf16* xq = (__bf16*)d_ws;                  // 1 MB bf16 copy of x

    const int n = BN * DN;
    xq_convert_kernel<<<(n + 255) / 256, 256, 0, stream>>>(x, xq, n);
    bilinear_wmma_kernel<<<dim3(UN, 2), 256, 0, stream>>>(x, W, V, bias, xq, out);
}
